// Actor_3032246911585
// MI455X (gfx1250) — matrix operand order repaired and checked by reading, not compiled, not run on hardware
//
#include <hip/hip_runtime.h>
#include <hip/hip_bf16.h>
#include <stdint.h>

typedef __attribute__((ext_vector_type(16))) __bf16 v16bf;
typedef __attribute__((ext_vector_type(8)))  __bf16 v8bf;
typedef __attribute__((ext_vector_type(8)))  float  v8f;

#define BN   2048   // batch
#define HN   256    // hidden
#define TN   64     // time
#define INN  7      // input features
#define G4N  1024   // 4*H

__device__ __forceinline__ float sigm(float x) { return 1.0f / (1.0f + __expf(-x)); }

// A-matrix frag (16-bit, 16x32): lanes 0-15 -> K 0..15, lanes 16-31 -> K 16..31 (contiguous)
__device__ __forceinline__ v16bf load_frag_A(const __bf16* A, int ld, int row, int kb, int lane) {
  const int chunk = (lane >> 4) << 4;                       // 0 or 16
  const __bf16* p = A + row * ld + kb + chunk;
  return *reinterpret_cast<const v16bf*>(p);
}

// B-matrix frag (16-bit, 32x16): lanes 0-15 -> K kb..kb+15, lanes 16-31 -> K kb+16..kb+31 (contiguous)
__device__ __forceinline__ v16bf load_frag_B(const __bf16* W, int ld, int nrow, int kb, int lane) {
  const int koff = (lane >> 4) << 4;                        // 0 or 16
  return *reinterpret_cast<const v16bf*>(W + nrow * ld + kb + koff);
}

__device__ __forceinline__ v8f zero8() {
  v8f z = {0.f,0.f,0.f,0.f,0.f,0.f,0.f,0.f};
  return z;
}

// ---------------------------------------------------------------------------
// Generic WMMA GEMM: out[m,n] = act( A[MxK](bf16) * W[NxK](bf16)^T + bias[n]*(rowscale?rowscale[m]:1) )
// Wave computes a 16(M) x 64(N) tile. M%16==0, N%64==0, K%32==0 assumed.
// ---------------------------------------------------------------------------
__global__ void gemm_wmma(const __bf16* __restrict__ A, const __bf16* __restrict__ W,
                          const float* __restrict__ bias, const float* __restrict__ rowscale,
                          float* __restrict__ outF, __bf16* __restrict__ outB, int ldOut,
                          int M, int N, int K, int act) {
  const int lane = threadIdx.x & 31;
  const int wid  = (blockIdx.x * blockDim.x + threadIdx.x) >> 5;
  const int ngrp = N >> 6;
  const int totalWaves = (M >> 4) * ngrp;
  if (wid >= totalWaves) return;
  const int mT = wid / ngrp;
  const int ng = wid % ngrp;
  const int rowA = mT * 16 + (lane & 15);

  v8f acc0 = zero8(), acc1 = zero8(), acc2 = zero8(), acc3 = zero8();
  for (int kb = 0; kb < K; kb += 32) {
    v16bf a = load_frag_A(A, K, rowA, kb, lane);
    v16bf b0 = load_frag_B(W, K, (ng << 6) +  0 + (lane & 15), kb, lane);
    v16bf b1 = load_frag_B(W, K, (ng << 6) + 16 + (lane & 15), kb, lane);
    v16bf b2 = load_frag_B(W, K, (ng << 6) + 32 + (lane & 15), kb, lane);
    v16bf b3 = load_frag_B(W, K, (ng << 6) + 48 + (lane & 15), kb, lane);
    acc0 = __builtin_amdgcn_wmma_f32_16x16x32_bf16(false, a, false, b0, (short)0, acc0, false, false);
    acc1 = __builtin_amdgcn_wmma_f32_16x16x32_bf16(false, a, false, b1, (short)0, acc1, false, false);
    acc2 = __builtin_amdgcn_wmma_f32_16x16x32_bf16(false, a, false, b2, (short)0, acc2, false, false);
    acc3 = __builtin_amdgcn_wmma_f32_16x16x32_bf16(false, a, false, b3, (short)0, acc3, false, false);
  }

  const int mBase = mT * 16 + ((lane >> 4) << 3);
  v8f accs[4] = {acc0, acc1, acc2, acc3};
#pragma unroll
  for (int s = 0; s < 4; ++s) {
    const int n = (ng << 6) + (s << 4) + (lane & 15);
    const float bv = bias ? bias[n] : 0.f;
#pragma unroll
    for (int v = 0; v < 8; ++v) {
      const int m = mBase + v;
      float val = accs[s][v] + (rowscale ? bv * rowscale[m] : bv);
      if (act) val = fmaxf(val, 0.f);
      if (outF) outF[m * ldOut + n] = val;
      if (outB) outB[m * ldOut + n] = (__bf16)val;
    }
  }
}

// ---------------------------------------------------------------------------
// Fused LSTM gate kernel: one wave owns a 16(b) x 16(h) patch and computes all
// four gate tiles {h, 256+h, 512+h, 768+h} so the nonlinearity stays in regs.
// MODE 0: serial step  (adds inline dot7(x, Wih_s[n]) + bias; c in-place; h -> bf16)
// MODE 1: mixing cell  (bias only; c0 in, c_m/h_m -> d_out f32, h_m -> bf16 ws)
// ---------------------------------------------------------------------------
template <int MODE>
__global__ void lstm_gates(const __bf16* __restrict__ A, const __bf16* __restrict__ W,
                           const float* __restrict__ bias, int K,
                           const float* __restrict__ state, const float* __restrict__ Wih, int tsrc,
                           const float* __restrict__ c_in, float* __restrict__ c_out,
                           __bf16* __restrict__ h_bf, float* __restrict__ h_f32) {
  const int lane = threadIdx.x & 31;
  const int wid  = (blockIdx.x * blockDim.x + threadIdx.x) >> 5;
  if (wid >= (BN / 16) * (HN / 16)) return;
  const int mT = wid >> 4;          // HN/16 == 16 column tiles
  const int hT = wid & 15;
  const int rowA = mT * 16 + (lane & 15);

  v8f acc[4] = {zero8(), zero8(), zero8(), zero8()};
  for (int kb = 0; kb < K; kb += 32) {
    v16bf a = load_frag_A(A, K, rowA, kb, lane);
#pragma unroll
    for (int g = 0; g < 4; ++g) {
      const int n = (g << 8) + (hT << 4) + (lane & 15);
      v16bf b = load_frag_B(W, K, n, kb, lane);
      acc[g] = __builtin_amdgcn_wmma_f32_16x16x32_bf16(false, a, false, b, (short)0, acc[g], false, false);
    }
  }

  const int hcol = (hT << 4) + (lane & 15);
  float bv[4];
  float wih[4][INN];
#pragma unroll
  for (int g = 0; g < 4; ++g) {
    const int n = (g << 8) + hcol;
    bv[g] = bias[n];
    if constexpr (MODE == 0) {
#pragma unroll
      for (int j = 0; j < INN; ++j) wih[g][j] = Wih[n * INN + j];
    }
  }

  const int mBase = mT * 16 + ((lane >> 4) << 3);
#pragma unroll
  for (int v = 0; v < 8; ++v) {
    const int m = mBase + v;
    float g4[4];
    if constexpr (MODE == 0) {
      const float* x = state + (m * TN + tsrc) * INN;
      float xv[INN];
#pragma unroll
      for (int j = 0; j < INN; ++j) xv[j] = x[j];
#pragma unroll
      for (int g = 0; g < 4; ++g) {
        float d = bv[g];
#pragma unroll
        for (int j = 0; j < INN; ++j) d += xv[j] * wih[g][j];
        g4[g] = acc[g][v] + d;
      }
    } else {
#pragma unroll
      for (int g = 0; g < 4; ++g) g4[g] = acc[g][v] + bv[g];
    }
    const float iv = sigm(g4[0]);
    const float fv = sigm(g4[1]);
    const float gv = tanhf(g4[2]);
    const float ov = sigm(g4[3]);
    const float cold = c_in[m * HN + hcol];
    const float c2 = fv * cold + iv * gv;
    const float h2 = ov * tanhf(c2);
    c_out[m * HN + hcol] = c2;
    h_bf[m * HN + hcol] = (__bf16)h2;
    if constexpr (MODE == 1) h_f32[m * HN + hcol] = h2;
  }
}

// ---------------------------------------------------------------------------
// Parallel-branch LSTM cell over all t>=1 with fused s0/s1-weighted T-reduction.
// One block per batch row b; thread h in [0,256). x rows staged in LDS.
// ---------------------------------------------------------------------------
__global__ void par_reduce(const float* __restrict__ state, const float* __restrict__ preG,
                           const float* __restrict__ Wih_p, const float* __restrict__ c0,
                           __bf16* __restrict__ w1o, __bf16* __restrict__ w2o,
                           float* __restrict__ s0sum, float* __restrict__ s1sum) {
  const int b = blockIdx.x;
  const int h = threadIdx.x;
  __shared__ float sx[63 * INN];
  const float* sb = state + b * TN * INN;
  for (int i = h; i < 63 * INN; i += 256) sx[i] = sb[INN + i];
  __syncthreads();

  float pre[4], wih[4][INN];
#pragma unroll
  for (int g = 0; g < 4; ++g) {
    const int n = (g << 8) + h;
    pre[g] = preG[b * G4N + n];
#pragma unroll
    for (int j = 0; j < INN; ++j) wih[g][j] = Wih_p[n * INN + j];
  }
  const float c0v = c0[b * HN + h];
  float w1 = 0.f, w2 = 0.f, s0s = 0.f, s1s = 0.f;
  for (int t = 0; t < 63; ++t) {
    const float* x = &sx[t * INN];
    float g4[4];
#pragma unroll
    for (int g = 0; g < 4; ++g) {
      float d = pre[g];
#pragma unroll
      for (int j = 0; j < INN; ++j) d += x[j] * wih[g][j];
      g4[g] = d;
    }
    const float c2 = sigm(g4[1]) * c0v + sigm(g4[0]) * tanhf(g4[2]);
    const float h2 = sigm(g4[3]) * tanhf(c2);
    const float s0 = x[0] * 0.1f;   // /LONG
    const float s1 = x[1];
    w1 += h2 * s0;
    w2 += h2 * s1;
    s0s += s0;
    s1s += s1;
  }
  w1o[b * HN + h] = (__bf16)w1;
  w2o[b * HN + h] = (__bf16)w2;
  if (h == 0) {                       // t=0 term contributes bias*s only (z masked to 0)
    s0sum[b] = s0s + sb[0] * 0.1f;
    s1sum[b] = s1s + sb[1];
  }
}

// Final tiny head: a[b,j] = tanh( relu_out[b,:] . Wl3[j,:] + bl3[j] ) * MAX_ACTION
__global__ void mlp_head(const __bf16* __restrict__ a1, const float* __restrict__ Wl3,
                         const float* __restrict__ bl3, float* __restrict__ out) {
  const int idx = blockIdx.x * blockDim.x + threadIdx.x;
  if (idx >= BN * 3) return;
  const int b = idx / 3, j = idx % 3;
  float s = bl3[j];
  const __bf16* ar = a1 + b * HN;
  const float* wr = Wl3 + j * HN;
  for (int k = 0; k < HN; ++k) s += (float)ar[k] * wr[k];
  out[idx] = tanhf(s) * 1.0f;
}

// -------------------------- prep kernels -----------------------------------
__global__ void cvt_bf16(const float* __restrict__ src, __bf16* __restrict__ dst,
                         int rows, int cols, int sld, int dld) {
  const int i = blockIdx.x * blockDim.x + threadIdx.x;
  if (i >= rows * cols) return;
  const int r = i / cols, c = i % cols;
  dst[r * dld + c] = (__bf16)src[r * sld + c];
}
__global__ void copy_bf16(const __bf16* __restrict__ src, __bf16* __restrict__ dst,
                          int rows, int cols, int sld, int dld) {
  const int i = blockIdx.x * blockDim.x + threadIdx.x;
  if (i >= rows * cols) return;
  const int r = i / cols, c = i % cols;
  dst[r * dld + c] = src[r * sld + c];
}
__global__ void vec_add(const float* __restrict__ a, const float* __restrict__ b,
                        float* __restrict__ d, int n) {
  const int i = blockIdx.x * blockDim.x + threadIdx.x;
  if (i < n) d[i] = a[i] + b[i];
}

// ---------------------------------------------------------------------------
extern "C" void kernel_launch(void* const* d_in, const int* in_sizes, int n_in,
                              void* d_out, int out_size, void* d_ws, size_t ws_size,
                              hipStream_t stream) {
  const float* state = (const float*)d_in[0];
  const float* h0    = (const float*)d_in[1];
  const float* c0    = (const float*)d_in[2];
  const float* Wih_s = (const float*)d_in[3];
  const float* Whh_s = (const float*)d_in[4];
  const float* bih_s = (const float*)d_in[5];
  const float* bhh_s = (const float*)d_in[6];
  const float* Wih_p = (const float*)d_in[7];
  const float* Whh_p = (const float*)d_in[8];
  const float* bih_p = (const float*)d_in[9];
  const float* bhh_p = (const float*)d_in[10];
  const float* W1    = (const float*)d_in[11];
  const float* b1    = (const float*)d_in[12];
  const float* W2    = (const float*)d_in[13];
  const float* b2    = (const float*)d_in[14];
  const float* Wih_m = (const float*)d_in[15];
  const float* Whh_m = (const float*)d_in[16];
  const float* bih_m = (const float*)d_in[17];
  const float* bhh_m = (const float*)d_in[18];
  const float* Wl2   = (const float*)d_in[19];
  const float* bl2   = (const float*)d_in[20];
  const float* Wl3   = (const float*)d_in[21];
  const float* bl3   = (const float*)d_in[22];

  float* outA = (float*)d_out;                       // [2048,1,3]
  float* outH = outA + BN * 3;                       // [1,2048,256]
  float* outC = outH + BN * HN;                      // [1,2048,256]

  // workspace carving (256B aligned)
  size_t off = 0;
  auto carve = [&](size_t bytes) {
    char* p = (char*)d_ws + off;
    off += (bytes + 255) & ~(size_t)255;
    return (void*)p;
  };
  __bf16* whh_s_bf = (__bf16*)carve(G4N * HN * 2);
  __bf16* whh_p_bf = (__bf16*)carve(G4N * HN * 2);
  __bf16* wcat_bf  = (__bf16*)carve(G4N * 768 * 2);   // [Wih_m | Whh_m]
  __bf16* w1_bf    = (__bf16*)carve(128 * HN * 2);
  __bf16* w2_bf    = (__bf16*)carve(128 * HN * 2);
  __bf16* wl2_bf   = (__bf16*)carve(HN * HN * 2);
  __bf16* hbuf0    = (__bf16*)carve(BN * HN * 2);
  __bf16* hbuf1    = (__bf16*)carve(BN * HN * 2);
  float*  c_ser    = (float*)carve(BN * HN * 4);
  float*  preG     = (float*)carve((size_t)BN * G4N * 4);
  __bf16* w1sum_bf = (__bf16*)carve(BN * HN * 2);
  __bf16* w2sum_bf = (__bf16*)carve(BN * HN * 2);
  float*  s0sum    = (float*)carve(BN * 4);
  float*  s1sum    = (float*)carve(BN * 4);
  __bf16* acat     = (__bf16*)carve((size_t)BN * 768 * 2);  // [h_ser | left | right | h0]
  __bf16* hm_bf    = (__bf16*)carve(BN * HN * 2);
  __bf16* a1_bf    = (__bf16*)carve(BN * HN * 2);
  float*  bias_s   = (float*)carve(G4N * 4);
  float*  bias_p   = (float*)carve(G4N * 4);
  float*  bias_m   = (float*)carve(G4N * 4);

  auto cvt = [&](const float* s, __bf16* d, int r, int c, int sld, int dld) {
    int n = r * c;
    cvt_bf16<<<(n + 255) / 256, 256, 0, stream>>>(s, d, r, c, sld, dld);
  };

  // weight / state conversions
  cvt(Whh_s, whh_s_bf, G4N, HN, HN, HN);
  cvt(Whh_p, whh_p_bf, G4N, HN, HN, HN);
  cvt(Wih_m, wcat_bf, G4N, 512, 512, 768);
  cvt(Whh_m, wcat_bf + 512, G4N, HN, HN, 768);
  cvt(W1, w1_bf, 128, HN, HN, HN);
  cvt(W2, w2_bf, 128, HN, HN, HN);
  cvt(Wl2, wl2_bf, HN, HN, HN, HN);
  cvt(h0, hbuf0, BN, HN, HN, HN);
  cvt(h0, acat + 512, BN, HN, HN, 768);
  vec_add<<<4, 256, 0, stream>>>(bih_s, bhh_s, bias_s, G4N);
  vec_add<<<4, 256, 0, stream>>>(bih_p, bhh_p, bias_p, G4N);
  vec_add<<<4, 256, 0, stream>>>(bih_m, bhh_m, bias_m, G4N);
  hipMemcpyAsync(c_ser, c0, (size_t)BN * HN * 4, hipMemcpyDeviceToDevice, stream);

  // parallel branch: preG = h0 @ Whh_p^T + (bih_p + bhh_p)   [2048 x 1024]
  {
    int waves = (BN / 16) * (G4N / 64);               // 2048
    gemm_wmma<<<waves / 8, 256, 0, stream>>>(hbuf0, whh_p_bf, bias_p, nullptr,
                                             preG, nullptr, G4N, BN, G4N, HN, 0);
  }
  par_reduce<<<BN, 256, 0, stream>>>(state, preG, Wih_p, c0, w1sum_bf, w2sum_bf, s0sum, s1sum);
  {
    int waves = (BN / 16) * (128 / 64);               // 256
    gemm_wmma<<<waves / 8, 256, 0, stream>>>(w1sum_bf, w1_bf, b1, s0sum,
                                             nullptr, acat + 256, 768, BN, 128, HN, 0);
    gemm_wmma<<<waves / 8, 256, 0, stream>>>(w2sum_bf, w2_bf, b2, s1sum,
                                             nullptr, acat + 384, 768, BN, 128, HN, 0);
  }

  // serial LSTM: 63 steps, step s consumes x = state[:, 63-s, :]
  __bf16* hp = hbuf0;
  __bf16* hn = hbuf1;
  for (int s = 0; s < 63; ++s) {
    lstm_gates<0><<<256, 256, 0, stream>>>(hp, whh_s_bf, bias_s, HN,
                                           state, Wih_s, 63 - s,
                                           c_ser, c_ser, hn, nullptr);
    __bf16* t = hp; hp = hn; hn = t;
  }
  // final h_ser now in hp; place into acat columns [0,256)
  copy_bf16<<<(BN * HN + 255) / 256, 256, 0, stream>>>(hp, acat, BN, HN, HN, 768);

  // mixing LSTM: K = 768 over [h_ser | par_hidden | h0]; writes h_m/c_m to d_out
  lstm_gates<1><<<256, 256, 0, stream>>>(acat, wcat_bf, bias_m, 768,
                                         nullptr, nullptr, 0,
                                         c0, outC, hm_bf, outH);

  // head: relu(h_m @ Wl2^T + bl2) -> a1 ; tanh(a1 @ Wl3^T + bl3)
  {
    int waves = (BN / 16) * (HN / 64);                // 512
    gemm_wmma<<<waves / 8, 256, 0, stream>>>(hm_bf, wl2_bf, bl2, nullptr,
                                             nullptr, a1_bf, HN, BN, HN, HN, 1);
  }
  mlp_head<<<(BN * 3 + 255) / 256, 256, 0, stream>>>(a1_bf, Wl3, bl3, outA);
}
